// Dense_4870492913834
// MI455X (gfx1250) — 8-bit matrix operand order repaired and checked by reading, compiled, not run on hardware
//
#include <hip/hip_runtime.h>

typedef __attribute__((ext_vector_type(16))) int   v16i;
typedef __attribute__((ext_vector_type(8)))  float v8f;

#define TOKENS 32768
#define DMODEL 1024
#define UNITS  4096

static __device__ __forceinline__ float clamp448(float v) {
  return fminf(fmaxf(v, -448.0f), 448.0f);
}

#if __has_builtin(__builtin_amdgcn_cvt_pk_fp8_f32)
static __device__ __forceinline__ unsigned pack_fp8x4(float a, float b, float c, float d) {
  int lo  = __builtin_amdgcn_cvt_pk_fp8_f32(clamp448(a), clamp448(b), 0,  false);
  int all = __builtin_amdgcn_cvt_pk_fp8_f32(clamp448(c), clamp448(d), lo, true);
  return (unsigned)all;
}
static __device__ __forceinline__ unsigned char fp8_byte(float a) {
  return (unsigned char)(__builtin_amdgcn_cvt_pk_fp8_f32(clamp448(a), 0.0f, 0, false) & 0xFF);
}
#else
static __device__ __forceinline__ unsigned char fp8_byte(float f) {
  f = clamp448(f);
  unsigned u    = __float_as_uint(f);
  unsigned sign = (u >> 24) & 0x80u;
  unsigned ax   = u & 0x7FFFFFFFu;
  if (ax == 0u) return (unsigned char)sign;
  if (ax < 0x3C800000u) {
    float a = __uint_as_float(ax);
    int m = (int)(a * 512.0f + 0.5f);
    if (m > 7) return (unsigned char)(sign | 0x08u);
    return (unsigned char)(sign | (unsigned)m);
  }
  unsigned r    = ax + (((ax >> 20) & 1u) + 0x7FFFFu);
  unsigned e    = (r >> 23) - 127u + 7u;
  unsigned code = (e << 3) | ((r >> 20) & 7u);
  if (code > 0x7Eu) code = 0x7Eu;
  return (unsigned char)(sign | code);
}
static __device__ __forceinline__ unsigned pack_fp8x4(float a, float b, float c, float d) {
  return (unsigned)fp8_byte(a) | ((unsigned)fp8_byte(b) << 8) |
         ((unsigned)fp8_byte(c) << 16) | ((unsigned)fp8_byte(d) << 24);
}
#endif

__global__ __launch_bounds__(256) void quant_x_kernel(const float* __restrict__ x,
                                                      unsigned* __restrict__ xq) {
  const int i = blockIdx.x * 256 + threadIdx.x;
  const float4 v = ((const float4*)x)[i];
  xq[i] = pack_fp8x4(v.x, v.y, v.z, v.w);
}

__global__ __launch_bounds__(256) void quant_wT_kernel(const float* __restrict__ w,
                                                       unsigned char* __restrict__ wT) {
  __shared__ unsigned char tile[64][80];
  const int k0 = blockIdx.x * 64;
  const int n0 = blockIdx.y * 64;
  const int t  = threadIdx.x;
#pragma unroll
  for (int i = 0; i < 16; ++i) {
    const int idx = i * 256 + t;
    const int kk  = idx >> 6;
    const int nn  = idx & 63;
    tile[nn][kk]  = fp8_byte(w[(size_t)(k0 + kk) * UNITS + (n0 + nn)]);
  }
  __syncthreads();
  const int nn = t >> 2;
  const int g  = t & 3;
  const uint4 d = *(const uint4*)&tile[nn][g * 16];
  *(uint4*)&wT[(size_t)(n0 + nn) * DMODEL + (k0 + g * 16)] = d;
}

__global__ __launch_bounds__(256) void fp8_gemm_bias_kernel(
    const unsigned char* __restrict__ Aq,
    const unsigned char* __restrict__ BqT,
    const float* __restrict__ bias,
    float* __restrict__ out) {
  const int lane = threadIdx.x & 31;
  const int wave = threadIdx.x >> 5;
  const int wm   = wave & 3;
  const int wn   = wave >> 2;
  const int l16  = lane & 15;
  const int lh   = lane >> 4;

  const int m0 = blockIdx.y * 128 + wm * 32;
  const int n0 = blockIdx.x * 128 + wn * 64;

  v8f acc[2][4] = {};

  const unsigned char* aRow0 = Aq + (size_t)(m0 + l16) * DMODEL + lh * 16;
  const unsigned char* aRow1 = aRow0 + (size_t)16 * DMODEL;
  const unsigned char* bCol[4];
#pragma unroll
  for (int ni = 0; ni < 4; ++ni)
    bCol[ni] = BqT + (size_t)(n0 + ni * 16 + l16) * DMODEL + lh * 16;

  for (int k0 = 0; k0 < DMODEL; k0 += 128) {
    v16i a[2], b[4];
#pragma unroll
    for (int p = 0; p < 8; ++p) {
      const uint2 d0 = *(const uint2*)(aRow0 + k0 + (p >> 1) * 32 + (p & 1) * 8);
      const uint2 d1 = *(const uint2*)(aRow1 + k0 + (p >> 1) * 32 + (p & 1) * 8);
      a[0][2 * p]     = (int)d0.x;
      a[0][2 * p + 1] = (int)d0.y;
      a[1][2 * p]     = (int)d1.x;
      a[1][2 * p + 1] = (int)d1.y;
    }
#pragma unroll
    for (int ni = 0; ni < 4; ++ni) {
#pragma unroll
      for (int g = 0; g < 4; ++g) {
        const uint4 d = *(const uint4*)(bCol[ni] + k0 + g * 32);
        b[ni][4 * g + 0] = (int)d.x;
        b[ni][4 * g + 1] = (int)d.y;
        b[ni][4 * g + 2] = (int)d.z;
        b[ni][4 * g + 3] = (int)d.w;
      }
    }
#pragma unroll
    for (int mi = 0; mi < 2; ++mi)
#pragma unroll
      for (int ni = 0; ni < 4; ++ni)
        acc[mi][ni] = __builtin_amdgcn_wmma_f32_16x16x128_fp8_fp8(
            a[mi], b[ni], (short)0, acc[mi][ni], false, false);
  }

#pragma unroll
  for (int ni = 0; ni < 4; ++ni) {
    const int   n  = n0 + ni * 16 + l16;
    const float bv = bias[n];
#pragma unroll
    for (int mi = 0; mi < 2; ++mi) {
#pragma unroll
      for (int r = 0; r < 8; ++r) {
        const int m = m0 + mi * 16 + lh * 8 + r;
        out[(size_t)m * UNITS + n] = acc[mi][ni][r] + bv;
      }
    }
  }
}

extern "C" void kernel_launch(void* const* d_in, const int* in_sizes, int n_in,
                              void* d_out, int out_size, void* d_ws, size_t ws_size,
                              hipStream_t stream) {
  const float* x    = (const float*)d_in[0];
  const float* w    = (const float*)d_in[1];
  const float* bias = (const float*)d_in[2];
  float* out = (float*)d_out;

  unsigned char* Aq  = (unsigned char*)d_ws;
  unsigned char* BqT = Aq + (size_t)TOKENS * DMODEL;

  quant_x_kernel<<<(TOKENS * DMODEL / 4) / 256, 256, 0, stream>>>(x, (unsigned*)Aq);

  dim3 gw(DMODEL / 64, UNITS / 64);
  quant_wT_kernel<<<gw, 256, 0, stream>>>(w, BqT);

  dim3 gg(UNITS / 128, TOKENS / 128);
  fp8_gemm_bias_kernel<<<gg, 256, 0, stream>>>(Aq, BqT, bias, out);
}
